// CausalSelfAttention_89292370084102
// MI455X (gfx1250) — matrix operand order repaired and checked by reading, not compiled, not run on hardware
//
#include <hip/hip_runtime.h>

typedef __attribute__((ext_vector_type(16))) __bf16 v16bf;
typedef __attribute__((ext_vector_type(8)))  __bf16 v8bf;
typedef __attribute__((ext_vector_type(8)))  float  v8f;
typedef __attribute__((__vector_size__(16))) int    v4i;

#define WMMA_BF16(a, b, c) \
  __builtin_amdgcn_wmma_f32_16x16x32_bf16(false, (a), false, (b), (short)0, (c), false, false)

// Problem constants
constexpr int Bn = 2;
constexpr int Sq = 2048;
constexpr int Dm = 1024;
constexpr int Hh = 16;
constexpr int HDd = 64;
constexpr int Mrows = Bn * Sq;       // 4096

// ---------------------------------------------------------------------------
// Async global -> LDS copy (gfx1250 path, ASYNCcnt-tracked), with fallback.
// ---------------------------------------------------------------------------
#if defined(__AMDGCN__) && __has_builtin(__builtin_amdgcn_global_load_async_to_lds_b128)
#define USE_ASYNC_LDS 1
#else
#define USE_ASYNC_LDS 0
#endif

__device__ __forceinline__ void cp_g2l_16B(const __bf16* g, __bf16* l) {
#if USE_ASYNC_LDS
  __builtin_amdgcn_global_load_async_to_lds_b128((v4i*)g, (v4i*)l, 0, 0);
#else
  *(v8bf*)l = *(const v8bf*)g;
#endif
}

__device__ __forceinline__ void wait_async_lds() {
#if USE_ASYNC_LDS
#if __has_builtin(__builtin_amdgcn_s_wait_asynccnt)
  __builtin_amdgcn_s_wait_asynccnt(0);
#else
  asm volatile("s_wait_asynccnt 0" ::: "memory");
#endif
#endif
}

// ---------------------------------------------------------------------------
// Stage 0: fp32 -> bf16 conversion kernels
// ---------------------------------------------------------------------------
__global__ __launch_bounds__(256) void cvt_bf16(const float* __restrict__ in,
                                                __bf16* __restrict__ out, int n) {
  int i = blockIdx.x * 256 + threadIdx.x;
  if (i < n) out[i] = (__bf16)in[i];
}

// out[n*rows + k] = in[k*cols + n]  (weight transpose: W[k,n] -> WT[n,k])
__global__ __launch_bounds__(256) void cvt_transpose_bf16(const float* __restrict__ in,
                                                          __bf16* __restrict__ out,
                                                          int rows, int cols) {
  int i = blockIdx.x * 256 + threadIdx.x;
  if (i < rows * cols) {
    int n = i / rows;
    int k = i - n * rows;
    out[i] = (__bf16)in[(size_t)k * cols + n];
  }
}

// ---------------------------------------------------------------------------
// WMMA GEMM: C[M,N] = A[M,K] (bf16 row-major) * WT[N,K]^T (bf16, pre-transposed)
// Block = 256 threads = 8 waves; wave tile = 16(M) x 64(N); block tile 128x64.
// B tile (64 x 32 bf16) staged in LDS per K-step, double-buffered, filled with
// async global->LDS (16B per thread = exactly one tile per block).
// MODE 0: bf16 row-major out; MODE 1: bf16 transposed-per-head out (Vt[b,h,hd,s]);
// MODE 2: fp32 row-major out.
// ---------------------------------------------------------------------------
constexpr int BSTRIDE = 40;   // padded row stride (bf16) -> conflict-free ds reads

template <int MODE>
__global__ __launch_bounds__(256) void gemm_wmma(const __bf16* __restrict__ A,
                                                 const __bf16* __restrict__ WT,
                                                 void* __restrict__ Out,
                                                 int M, int N, int K) {
  __shared__ __bf16 Bt[2][64 * BSTRIDE];

  const int tid  = threadIdx.x;
  const int lane = tid & 31;
  const int wave = tid >> 5;
  const int hi   = lane >> 4;
  const int lo   = lane & 15;
  const int m0   = blockIdx.y * 128 + wave * 16;
  const int n0   = blockIdx.x * 64;

  // Cooperative B-tile staging: thread -> (n_local = tid/4, k_local = (tid%4)*8)
  const int nl = tid >> 2;
  const int kl = (tid & 3) * 8;
  const __bf16* gB = WT + (size_t)(n0 + nl) * K + kl;
  __bf16* lB0 = &Bt[0][nl * BSTRIDE + kl];
  __bf16* lB1 = &Bt[1][nl * BSTRIDE + kl];

  v8f acc[4] = {v8f{}, v8f{}, v8f{}, v8f{}};
  const __bf16* Arow = A + (size_t)(m0 + lo) * K;

  // Prologue: stage K-step 0 into buffer 0.
  cp_g2l_16B(gB, lB0);
  wait_async_lds();
  __syncthreads();

  for (int kk = 0; kk < K; kk += 32) {
    const int cur = (kk >> 5) & 1;
    // Prefetch next K-step into the other buffer (overlaps with WMMAs below).
    if (kk + 32 < K) cp_g2l_16B(gB + kk + 32, cur ? lB0 : lB1);

    v16bf af;
    ((v8bf*)&af)[0] = *(const v8bf*)(Arow + kk + hi * 16);       // K = hi*16 .. +7
    ((v8bf*)&af)[1] = *(const v8bf*)(Arow + kk + hi * 16 + 8);   // K = hi*16+8 .. +15
#pragma unroll
    for (int t = 0; t < 4; ++t) {
      const __bf16* lrow = &Bt[cur][(t * 16 + lo) * BSTRIDE + hi * 16];
      v16bf bfr;
      ((v8bf*)&bfr)[0] = *(const v8bf*)(lrow);                   // K = hi*16 .. +7
      ((v8bf*)&bfr)[1] = *(const v8bf*)(lrow + 8);               // K = hi*16+8 .. +15
      acc[t] = WMMA_BF16(af, bfr, acc[t]);
    }

    wait_async_lds();
    __syncthreads();   // next buffer filled; everyone done reading cur
  }

#pragma unroll
  for (int t = 0; t < 4; ++t) {
#pragma unroll
    for (int r = 0; r < 8; ++r) {
      const int m = m0 + r + 8 * hi;
      const int n = n0 + t * 16 + lo;
      if (MODE == 0) {
        ((__bf16*)Out)[(size_t)m * N + n] = (__bf16)acc[t][r];
      } else if (MODE == 1) {
        const int b = m >> 11;            // m / Sq
        const int s = m & (Sq - 1);
        const int h = n >> 6;             // n / 64
        const int hd = n & 63;
        ((__bf16*)Out)[(((size_t)b * Hh + h) * HDd + hd) * Sq + s] = (__bf16)acc[t][r];
      } else {
        ((float*)Out)[(size_t)m * N + n] = acc[t][r];
      }
    }
  }
}

// ---------------------------------------------------------------------------
// Flash attention: one wave = 32 query rows x one head (two 16-row M-frags
// sharing every K/V fragment). bf16 WMMA, fp32 online softmax.
// Q/K bf16 row-major [B*S, D]; Vt bf16 [B,H,HD,S]; out bf16 row-major [B*S, D].
// ---------------------------------------------------------------------------
__global__ __launch_bounds__(128) void attn_wmma(const __bf16* __restrict__ Q,
                                                 const __bf16* __restrict__ Kb,
                                                 const __bf16* __restrict__ Vt,
                                                 __bf16* __restrict__ Ao) {
  __shared__ float Pl[4][32][33];     // per-wave P tile (32 q-rows x 32 keys, padded)

  const int wave = threadIdx.x >> 5;
  const int lane = threadIdx.x & 31;
  const int hi   = lane >> 4;
  const int lo   = lane & 15;

  const int w  = blockIdx.x * 4 + wave;
  const int qt = w & 63;              // S/32 = 64 query tiles
  const int h  = (w >> 6) & 15;
  const int b  = w >> 10;
  const int q0 = qt * 32;

  // Load Q A-fragments once: 2 row tiles x (K-dim 64 split into 2x32)
  v16bf qa[2][2];
#pragma unroll
  for (int u = 0; u < 2; ++u) {
    const __bf16* qptr = Q + (size_t)(b * Sq + q0 + u * 16 + lo) * Dm + h * HDd;
    ((v8bf*)&qa[u][0])[0] = *(const v8bf*)(qptr + hi * 16);
    ((v8bf*)&qa[u][0])[1] = *(const v8bf*)(qptr + hi * 16 + 8);
    ((v8bf*)&qa[u][1])[0] = *(const v8bf*)(qptr + 32 + hi * 16);
    ((v8bf*)&qa[u][1])[1] = *(const v8bf*)(qptr + 32 + hi * 16 + 8);
  }

  float mrow[2][8], lrow[2][8];
#pragma unroll
  for (int u = 0; u < 2; ++u)
#pragma unroll
    for (int r = 0; r < 8; ++r) { mrow[u][r] = -1.0e30f; lrow[u][r] = 0.0f; }
  v8f o[2][4] = {{v8f{}, v8f{}, v8f{}, v8f{}}, {v8f{}, v8f{}, v8f{}, v8f{}}};

  const float scale = 0.125f;         // 1/sqrt(64)

  for (int k0 = 0; k0 <= q0 + 31; k0 += 32) {
    // ---- scores: shared K fragments feed both row tiles (8 WMMAs) ----
    v8f sfr[2][2];
#pragma unroll
    for (int n2 = 0; n2 < 2; ++n2) {
      const int key = k0 + n2 * 16 + lo;          // <= q0+31 <= 2047
      const __bf16* kptr = Kb + (size_t)(b * Sq + key) * Dm + h * HDd;
      v16bf kb0, kb1;
      ((v8bf*)&kb0)[0] = *(const v8bf*)(kptr + hi * 16);
      ((v8bf*)&kb0)[1] = *(const v8bf*)(kptr + hi * 16 + 8);
      ((v8bf*)&kb1)[0] = *(const v8bf*)(kptr + 32 + hi * 16);
      ((v8bf*)&kb1)[1] = *(const v8bf*)(kptr + 32 + hi * 16 + 8);
#pragma unroll
      for (int u = 0; u < 2; ++u) {
        v8f s = v8f{};
        s = WMMA_BF16(qa[u][0], kb0, s);
        s = WMMA_BF16(qa[u][1], kb1, s);
        sfr[u][n2] = s;
      }
    }

    // ---- scale + causal mask + online softmax per row tile ----
    __builtin_amdgcn_wave_barrier();
#pragma unroll
    for (int u = 0; u < 2; ++u) {
      const bool needmask = (k0 + 31 > q0 + u * 16);
      float sc[2][8];
#pragma unroll
      for (int n2 = 0; n2 < 2; ++n2) {
#pragma unroll
        for (int r = 0; r < 8; ++r) {
          float v = sfr[u][n2][r] * scale;
          if (needmask) {
            const int query = q0 + u * 16 + r + 8 * hi;
            const int key   = k0 + n2 * 16 + lo;
            if (key > query) v = -__builtin_inff();
          }
          sc[n2][r] = v;
        }
      }
#pragma unroll
      for (int r = 0; r < 8; ++r) {
        float v = fmaxf(sc[0][r], sc[1][r]);
        v = fmaxf(v, __shfl_xor(v, 1));
        v = fmaxf(v, __shfl_xor(v, 2));
        v = fmaxf(v, __shfl_xor(v, 4));
        v = fmaxf(v, __shfl_xor(v, 8));
        const float mnew = fmaxf(mrow[u][r], v);
        const float corr = __expf(mrow[u][r] - mnew);
        mrow[u][r] = mnew;
        const float p0 = __expf(sc[0][r] - mnew);
        const float p1 = __expf(sc[1][r] - mnew);
        float rs = p0 + p1;
        rs += __shfl_xor(rs, 1);
        rs += __shfl_xor(rs, 2);
        rs += __shfl_xor(rs, 4);
        rs += __shfl_xor(rs, 8);
        lrow[u][r] = lrow[u][r] * corr + rs;
#pragma unroll
        for (int t = 0; t < 4; ++t) o[u][t][r] *= corr;
        const int M = u * 16 + r + 8 * hi;
        Pl[wave][M][lo]      = p0;
        Pl[wave][M][16 + lo] = p1;
      }
    }
    __builtin_amdgcn_wave_barrier();

    // ---- re-fragment P from LDS into A-layout (two 16x32 fragments) ----
    v16bf pa[2];
#pragma unroll
    for (int u = 0; u < 2; ++u) {
      const float* prow = &Pl[wave][u * 16 + lo][0];
#pragma unroll
      for (int j = 0; j < 8; ++j) pa[u][j] = (__bf16)prow[hi * 16 + j];
#pragma unroll
      for (int j = 0; j < 8; ++j) pa[u][8 + j] = (__bf16)prow[hi * 16 + 8 + j];
    }
    __builtin_amdgcn_wave_barrier();

    // ---- PV: shared V fragments feed both row tiles (8 WMMAs) ----
#pragma unroll
    for (int t = 0; t < 4; ++t) {
      const int hd = t * 16 + lo;
      const __bf16* vptr = Vt + (((size_t)(b * Hh + h)) * HDd + hd) * Sq + k0 + hi * 16;
      v16bf vb;
      ((v8bf*)&vb)[0] = *(const v8bf*)(vptr);
      ((v8bf*)&vb)[1] = *(const v8bf*)(vptr + 8);
#pragma unroll
      for (int u = 0; u < 2; ++u) o[u][t] = WMMA_BF16(pa[u], vb, o[u][t]);
    }
  }

  // ---- normalize and store attended (bf16 row-major [B*S, D]) ----
#pragma unroll
  for (int u = 0; u < 2; ++u) {
#pragma unroll
    for (int t = 0; t < 4; ++t) {
#pragma unroll
      for (int r = 0; r < 8; ++r) {
        const int M = u * 16 + r + 8 * hi;
        const float val = o[u][t][r] / lrow[u][r];
        Ao[(size_t)(b * Sq + q0 + M) * Dm + h * HDd + t * 16 + lo] = (__bf16)val;
      }
    }
  }
}

// ---------------------------------------------------------------------------
// Host launcher
// ---------------------------------------------------------------------------
extern "C" void kernel_launch(void* const* d_in, const int* in_sizes, int n_in,
                              void* d_out, int out_size, void* d_ws, size_t ws_size,
                              hipStream_t stream) {
  const float* x  = (const float*)d_in[0];
  const float* Wq = (const float*)d_in[1];
  const float* Wk = (const float*)d_in[2];
  const float* Wv = (const float*)d_in[3];
  const float* Wo = (const float*)d_in[4];

  char* ws = (char*)d_ws;
  size_t off = 0;
  auto alloc = [&](size_t bytes) { char* p = ws + off; off += bytes; return p; };

  const size_t xElems = (size_t)Mrows * Dm;       // 4M
  const size_t wElems = (size_t)Dm * Dm;          // 1M

  __bf16* xbf = (__bf16*)alloc(xElems * 2);
  __bf16* WqT = (__bf16*)alloc(wElems * 2);
  __bf16* WkT = (__bf16*)alloc(wElems * 2);
  __bf16* WvT = (__bf16*)alloc(wElems * 2);
  __bf16* WoT = (__bf16*)alloc(wElems * 2);
  __bf16* Qbf = (__bf16*)alloc(xElems * 2);
  __bf16* Kbf = (__bf16*)alloc(xElems * 2);
  __bf16* Vt  = (__bf16*)alloc(xElems * 2);
  __bf16* Abf = (__bf16*)alloc(xElems * 2);

  // Stage 0: conversions
  cvt_bf16<<<(int)(xElems / 256), 256, 0, stream>>>(x, xbf, (int)xElems);
  cvt_transpose_bf16<<<(int)(wElems / 256), 256, 0, stream>>>(Wq, WqT, Dm, Dm);
  cvt_transpose_bf16<<<(int)(wElems / 256), 256, 0, stream>>>(Wk, WkT, Dm, Dm);
  cvt_transpose_bf16<<<(int)(wElems / 256), 256, 0, stream>>>(Wv, WvT, Dm, Dm);
  cvt_transpose_bf16<<<(int)(wElems / 256), 256, 0, stream>>>(Wo, WoT, Dm, Dm);

  // Stage 1: projections
  dim3 gg(Dm / 64, Mrows / 128);   // (16, 32)
  gemm_wmma<0><<<gg, 256, 0, stream>>>(xbf, WqT, Qbf, Mrows, Dm, Dm);
  gemm_wmma<0><<<gg, 256, 0, stream>>>(xbf, WkT, Kbf, Mrows, Dm, Dm);
  gemm_wmma<1><<<gg, 256, 0, stream>>>(xbf, WvT, Vt,  Mrows, Dm, Dm);

  // Stage 2: flash attention (B*H*(S/32) waves / 4 waves per block)
  attn_wmma<<<(Bn * Hh * (Sq / 32)) / 4, 128, 0, stream>>>(Qbf, Kbf, Vt, Abf);

  // Stage 3: output projection -> fp32
  gemm_wmma<2><<<gg, 256, 0, stream>>>(Abf, WoT, d_out, Mrows, Dm, Dm);

  (void)in_sizes; (void)n_in; (void)out_size; (void)ws_size;
}
